// SCAESuite_43035572306299
// MI455X (gfx1250) — matrix operand order repaired and checked by reading, not compiled, not run on hardware
//
#include <hip/hip_runtime.h>
#include <stdint.h>

#define DEVINL __device__ __forceinline__

typedef __attribute__((ext_vector_type(16))) __bf16 v16bf;
typedef __attribute__((ext_vector_type(8)))  float  v8f;

constexpr int Dk   = 768;     // model dim
constexpr int Ftot = 12288;   // feature dim (up and down)
constexpr int Ntok = 4096;    // B*S
constexpr int KTOP = 64;      // k_topk (fixed by the reference setup)
constexpr int ELLW = 512;     // padded row width for compacted maskedT (avg nnz ~123)

union FragU { v16bf v; unsigned short u[16]; };

DEVINL unsigned short f2bf(float x) {
  unsigned int u = __float_as_uint(x);
  unsigned int r = u + 0x7FFFu + ((u >> 16) & 1u);   // round-to-nearest-even
  return (unsigned short)(r >> 16);
}
DEVINL float bf2f(unsigned short h) {
  return __uint_as_float(((unsigned int)h) << 16);
}

// ---- optional CDNA5 async global->LDS copy (guarded; sync fallback) -------
#if defined(__has_builtin)
#if __has_builtin(__builtin_amdgcn_global_load_async_to_lds_b128) && \
    __has_builtin(__builtin_amdgcn_s_wait_asynccnt)
#define USE_ASYNC_LDS 1
#endif
#endif
#ifndef USE_ASYNC_LDS
#define USE_ASYNC_LDS 0
#endif

#if USE_ASYNC_LDS
typedef int v4i_t __attribute__((ext_vector_type(4)));
typedef __attribute__((address_space(1))) v4i_t* g_v4i_p;   // global (AS1)
typedef __attribute__((address_space(3))) v4i_t* l_v4i_p;   // LDS (AS3)
DEVINL void async_copy_b128(const void* g, void* l) {
  // integer round-trip builds correctly typed AS pointers; generic LDS addr
  // truncates to the 32-bit LDS offset (ISA 10.2 aperture rules).
  __builtin_amdgcn_global_load_async_to_lds_b128(
      (g_v4i_p)(unsigned long long)g,
      (l_v4i_p)(unsigned int)(unsigned long long)l, 0, 0);
}
#endif

// ---------------------------------------------------------------------------
// WMMA fragment loaders (bf16, 16x16x32, wave32 layouts per ISA 7.12.2)
// ---------------------------------------------------------------------------
DEVINL v16bf load_frag_a(const unsigned short* __restrict__ A, int row0, int k0, int lane) {
  FragU f;
  const unsigned short* r =
      A + (size_t)(row0 + (lane & 15)) * Dk + k0 + ((lane >> 4) << 3);
#pragma unroll
  for (int e = 0; e < 8; ++e) f.u[e] = r[e];
#pragma unroll
  for (int e = 0; e < 8; ++e) f.u[8 + e] = r[16 + e];
  return f.v;
}

DEVINL v16bf load_frag_b(const unsigned short* __restrict__ Bt, int col0, int k0, int lane) {
  FragU f;
  const unsigned short* r =
      Bt + (size_t)(col0 + (lane & 15)) * Dk + k0 + ((lane >> 4) << 3);
#pragma unroll
  for (int e = 0; e < 8; ++e) f.u[e] = r[e];
#pragma unroll
  for (int e = 0; e < 8; ++e) f.u[8 + e] = r[16 + e];
  return f.v;
}

// ---------------------------------------------------------------------------
// GEMM: C[M,N] = A[M,K] * Bt[N,K]^T, K = Dk, bf16 operands row-major over K.
// MODE 0: Cf = relu(C + bias[n])            (up_pre)
// MODE 1: Cf = C                            (initial)
// MODE 2: Cbf = mask[n*F+m] ? bf16(C) : 0   (maskedT = (conn_mask .* virtual)^T)
// Block = 256 threads = 8 wave32 (2x4); wave tile 64x32; block tile 128x128.
// ---------------------------------------------------------------------------
template <int MODE>
__global__ __launch_bounds__(256) void wmma_gemm_kernel(
    const unsigned short* __restrict__ A,
    const unsigned short* __restrict__ Bt,
    float* __restrict__ Cf,
    unsigned short* __restrict__ Cbf,
    const float* __restrict__ bias,
    const unsigned char* __restrict__ mask) {
  const int lane = threadIdx.x & 31;
  const int wave = threadIdx.x >> 5;
  const int wm = wave >> 2;  // 0..1
  const int wn = wave & 3;   // 0..3
  const int baseM = blockIdx.y * 128 + wm * 64;
  const int baseN = blockIdx.x * 128 + wn * 32;

  v8f acc[4][2];
#pragma unroll
  for (int i = 0; i < 4; ++i)
#pragma unroll
    for (int j = 0; j < 2; ++j) acc[i][j] = (v8f){0, 0, 0, 0, 0, 0, 0, 0};

  for (int k0 = 0; k0 < Dk; k0 += 32) {
    if (k0 + 32 < Dk) {  // next K-tile hint (global_prefetch_b8)
      __builtin_prefetch(A + (size_t)(baseM + (lane & 15)) * Dk + k0 + 32, 0, 1);
      __builtin_prefetch(Bt + (size_t)(baseN + (lane & 15)) * Dk + k0 + 32, 0, 1);
    }
    v16bf fa[4], fb[2];
#pragma unroll
    for (int i = 0; i < 4; ++i) fa[i] = load_frag_a(A, baseM + 16 * i, k0, lane);
#pragma unroll
    for (int j = 0; j < 2; ++j) fb[j] = load_frag_b(Bt, baseN + 16 * j, k0, lane);
#pragma unroll
    for (int i = 0; i < 4; ++i)
#pragma unroll
      for (int j = 0; j < 2; ++j)
        acc[i][j] = __builtin_amdgcn_wmma_f32_16x16x32_bf16(
            false, fa[i], false, fb[j], (short)0, acc[i][j], false, false);
  }

  const int nlo = lane & 15;
  const int mhi = (lane >> 4) * 8;
#pragma unroll
  for (int i = 0; i < 4; ++i) {
#pragma unroll
    for (int j = 0; j < 2; ++j) {
      const int n = baseN + 16 * j + nlo;
#pragma unroll
      for (int r = 0; r < 8; ++r) {
        const int m = baseM + 16 * i + mhi + r;
        float v = acc[i][j][r];
        if (MODE == 0) {
          v += bias[n];
          v = v > 0.f ? v : 0.f;
          Cf[(size_t)m * Ftot + n] = v;
        } else if (MODE == 1) {
          Cf[(size_t)m * Ftot + n] = v;
        } else {
          unsigned short o = mask[(size_t)n * Ftot + m] ? f2bf(v) : (unsigned short)0;
          Cbf[(size_t)m * Ftot + n] = o;
        }
      }
    }
  }
}

// ---------------------------------------------------------------------------
// Compact each maskedT row into packed ELL entries (idx<<16 | bf16val).
// One thread per row, sequential scan -> fully deterministic entry order.
// Dropped bf16-zero entries contribute exactly 0 -> no numeric change.
// ---------------------------------------------------------------------------
__global__ void compact_kernel(const unsigned short* __restrict__ maskedT,
                               unsigned int* __restrict__ ell,
                               int* __restrict__ counts) {
  int j = blockIdx.x * blockDim.x + threadIdx.x;
  if (j >= Ftot) return;
  const unsigned short* row = maskedT + (size_t)j * Ftot;
  unsigned int* erow = ell + (size_t)j * ELLW;
  int cnt = 0;
  for (int i = 0; i < Ftot; ++i) {
    unsigned short h = row[i];
    if (h != 0) {
      if (cnt < ELLW) erow[cnt] = ((unsigned int)i << 16) | (unsigned int)h;
      ++cnt;
    }
  }
  counts[j] = cnt < ELLW ? cnt : ELLW;
}

// ---------------------------------------------------------------------------
// Top-K per row (up path): row staged in LDS (48KB of 320KB), 64 argmax
// reduction iterations (ties -> lowest index). Async copy when available.
// ---------------------------------------------------------------------------
__global__ __launch_bounds__(256) void topk_kernel(
    const float* __restrict__ rows, int* __restrict__ out_idx,
    float* __restrict__ out_val) {
  __shared__ float sv[Ftot];
  __shared__ float rmax[256];
  __shared__ int rloc[256];
  const int tid = threadIdx.x;
  const float* row = rows + (size_t)blockIdx.x * Ftot;
#if USE_ASYNC_LDS
  for (int i = tid * 4; i < Ftot; i += 256 * 4) async_copy_b128(row + i, &sv[i]);
  __builtin_amdgcn_s_wait_asynccnt(0);
  __syncthreads();
#else
  for (int i = tid; i < Ftot; i += 256) sv[i] = row[i];
  __syncthreads();
#endif
  for (int it = 0; it < KTOP; ++it) {
    float bv = -INFINITY;
    int bi = Ftot;
    for (int i = tid; i < Ftot; i += 256) {
      float v = sv[i];
      if (v > bv) { bv = v; bi = i; }
    }
    rmax[tid] = bv;
    rloc[tid] = bi;
    __syncthreads();
    for (int s = 128; s > 0; s >>= 1) {
      if (tid < s) {
        float ov = rmax[tid + s];
        int oi = rloc[tid + s];
        if (ov > rmax[tid] || (ov == rmax[tid] && oi < rloc[tid])) {
          rmax[tid] = ov;
          rloc[tid] = oi;
        }
      }
      __syncthreads();
    }
    if (tid == 0) {
      int w = rloc[0];
      out_idx[(size_t)blockIdx.x * KTOP + it] = w;
      out_val[(size_t)blockIdx.x * KTOP + it] = rmax[0];
      sv[w] = -INFINITY;
    }
    __syncthreads();
  }
}

// ---------------------------------------------------------------------------
// Fused down path (one workgroup per token):
//   arow = initial + bias_contrib + b_enc_down              (LDS, 48KB)
//   for each of 64 active up features: scatter its ELL row into arow
//     (phases separated by barriers; indices within a phase are distinct ->
//      no atomics needed -> bitwise deterministic)
//   top-64 of arow, then decode 64 rows of WddT into recon[t,:].
// ---------------------------------------------------------------------------
__global__ __launch_bounds__(256) void down_fused_kernel(
    const float* __restrict__ initial, const unsigned int* __restrict__ ell,
    const int* __restrict__ counts, const int* __restrict__ up_idx,
    const float* __restrict__ up_val, const float* __restrict__ bias_contrib,
    const float* __restrict__ b_enc_down,
    const unsigned short* __restrict__ WddT,
    const float* __restrict__ b_dec_down, float* __restrict__ out) {
  __shared__ float arow[Ftot];
  __shared__ int sj[KTOP];
  __shared__ float svv[KTOP];
  __shared__ int scnt[KTOP];
  __shared__ float rmax[256];
  __shared__ int rloc[256];
  __shared__ int dn_i[KTOP];
  __shared__ float dn_v[KTOP];
  const int t = blockIdx.x;
  const int tid = threadIdx.x;
  if (tid < KTOP) {
    int j = up_idx[(size_t)t * KTOP + tid];
    sj[tid] = j;
    svv[tid] = up_val[(size_t)t * KTOP + tid];
    scnt[tid] = counts[j];
  }
  const float* irow = initial + (size_t)t * Ftot;
  for (int i = tid; i < Ftot; i += 256)
    arow[i] = irow[i] + bias_contrib[i] + b_enc_down[i];
  __syncthreads();

  for (int jj = 0; jj < KTOP; ++jj) {
    const unsigned int* erow = ell + (size_t)sj[jj] * ELLW;
    const float vj = svv[jj];
    const int c = scnt[jj];
    for (int e = tid; e < c; e += 256) {
      unsigned int u = erow[e];
      arow[u >> 16] += vj * bf2f((unsigned short)(u & 0xFFFFu));
    }
    __syncthreads();
  }

  // top-64 of approx row
  for (int it = 0; it < KTOP; ++it) {
    float bv = -INFINITY;
    int bi = Ftot;
    for (int i = tid; i < Ftot; i += 256) {
      float v = arow[i];
      if (v > bv) { bv = v; bi = i; }
    }
    rmax[tid] = bv;
    rloc[tid] = bi;
    __syncthreads();
    for (int s = 128; s > 0; s >>= 1) {
      if (tid < s) {
        float ov = rmax[tid + s];
        int oi = rloc[tid + s];
        if (ov > rmax[tid] || (ov == rmax[tid] && oi < rloc[tid])) {
          rmax[tid] = ov;
          rloc[tid] = oi;
        }
      }
      __syncthreads();
    }
    if (tid == 0) {
      int w = rloc[0];
      dn_i[it] = w;
      dn_v[it] = rmax[0];
      arow[w] = -INFINITY;
    }
    __syncthreads();
  }

  // sparse decode: recon[t,d] = b_dec_down[d] + sum_j dn_v[j]*WddT[dn_i[j],d]
  for (int d = tid; d < Dk; d += 256) {
    float a = b_dec_down[d];
    for (int jj = 0; jj < KTOP; ++jj)
      a += dn_v[jj] * bf2f(WddT[(size_t)dn_i[jj] * Dk + d]);
    out[(size_t)t * Dk + d] = a;
  }
}

// bias_contrib[i] = b_dec_up . W_enc_down[i,:]
__global__ void biasc_kernel(const float* __restrict__ b_dec_up,
                             const float* __restrict__ W_enc_down,
                             float* __restrict__ outv) {
  int i = blockIdx.x * blockDim.x + threadIdx.x;
  if (i >= Ftot) return;
  const float* r = W_enc_down + (size_t)i * Dk;
  float a = 0.f;
  for (int d = 0; d < Dk; ++d) a += b_dec_up[d] * r[d];
  outv[i] = a;
}

// ---- conversion kernels --------------------------------------------------
__global__ void f2bf_kernel(const float* __restrict__ src,
                            unsigned short* __restrict__ dst, size_t n) {
  size_t i = (size_t)blockIdx.x * blockDim.x + threadIdx.x;
  size_t stride = (size_t)gridDim.x * blockDim.x;
  for (; i < n; i += stride) dst[i] = f2bf(src[i]);
}

// src [R,C] row-major  ->  dst [C,R] row-major (transpose + bf16)
__global__ void f2bfT_kernel(const float* __restrict__ src,
                             unsigned short* __restrict__ dst, int R, int C) {
  size_t n = (size_t)R * C;
  size_t i = (size_t)blockIdx.x * blockDim.x + threadIdx.x;
  size_t stride = (size_t)gridDim.x * blockDim.x;
  for (; i < n; i += stride) {
    int c = (int)(i / R);
    int r = (int)(i % R);
    dst[i] = f2bf(src[(size_t)r * C + c]);
  }
}

// dst[t*Dk+d] = bf16(x[t*Dk+d] - b[d])
__global__ void sub_bf_kernel(const float* __restrict__ x,
                              const float* __restrict__ b,
                              unsigned short* __restrict__ dst, size_t n) {
  size_t i = (size_t)blockIdx.x * blockDim.x + threadIdx.x;
  size_t stride = (size_t)gridDim.x * blockDim.x;
  for (; i < n; i += stride) dst[i] = f2bf(x[i] - b[i % Dk]);
}

// ---------------------------------------------------------------------------
extern "C" void kernel_launch(void* const* d_in, const int* in_sizes, int n_in,
                              void* d_out, int out_size, void* d_ws,
                              size_t ws_size, hipStream_t stream) {
  (void)in_sizes; (void)n_in; (void)out_size; (void)ws_size;
  const float* x_up       = (const float*)d_in[0];
  const float* x_resid    = (const float*)d_in[1];
  const float* W_enc_up   = (const float*)d_in[2];
  const float* b_enc_up   = (const float*)d_in[3];
  const float* b_dec_up   = (const float*)d_in[4];
  const float* W_dec_up   = (const float*)d_in[5];   // [D,F]
  const float* W_enc_down = (const float*)d_in[6];
  const float* b_enc_down = (const float*)d_in[7];
  const float* b_dec_down = (const float*)d_in[8];
  const float* W_dec_down = (const float*)d_in[9];   // [D,F]
  const unsigned char* conn_mask = (const unsigned char*)d_in[10];  // [F,F] bool
  // d_in[11] = k_topk scalar == 64 (compile-time KTOP)

  char* ws = (char*)d_ws;
  size_t off = 0;
  auto bump = [&](size_t bytes) -> char* {
    char* p = ws + off;
    off = (off + bytes + 255) & ~(size_t)255;
    return p;
  };

  unsigned short* Wup_bf = (unsigned short*)bump((size_t)Ftot * Dk * 2);
  unsigned short* Wdn_bf = (unsigned short*)bump((size_t)Ftot * Dk * 2);
  unsigned short* WdupT  = (unsigned short*)bump((size_t)Ftot * Dk * 2);
  unsigned short* WddT   = (unsigned short*)bump((size_t)Ftot * Dk * 2);
  unsigned short* Xc     = (unsigned short*)bump((size_t)Ntok * Dk * 2);
  unsigned short* Xr     = (unsigned short*)bump((size_t)Ntok * Dk * 2);
  float* up_pre          = (float*)bump((size_t)Ntok * Ftot * 4);
  float* initial         = (float*)bump((size_t)Ntok * Ftot * 4);
  unsigned short* maskedT= (unsigned short*)bump((size_t)Ftot * Ftot * 2);
  unsigned int* ell      = (unsigned int*)bump((size_t)Ftot * ELLW * 4);
  int* counts            = (int*)bump((size_t)Ftot * 4);
  float* biasc           = (float*)bump((size_t)Ftot * 4);
  int*   up_idx          = (int*)bump((size_t)Ntok * KTOP * 4);
  float* up_val          = (float*)bump((size_t)Ntok * KTOP * 4);

  const dim3 eb(256);
  // --- bf16 operand prep ---
  f2bf_kernel<<<dim3(2048), eb, 0, stream>>>(W_enc_up, Wup_bf, (size_t)Ftot * Dk);
  f2bf_kernel<<<dim3(2048), eb, 0, stream>>>(W_enc_down, Wdn_bf, (size_t)Ftot * Dk);
  f2bfT_kernel<<<dim3(2048), eb, 0, stream>>>(W_dec_up, WdupT, Dk, Ftot);
  f2bfT_kernel<<<dim3(2048), eb, 0, stream>>>(W_dec_down, WddT, Dk, Ftot);
  sub_bf_kernel<<<dim3(2048), eb, 0, stream>>>(x_up, b_dec_up, Xc, (size_t)Ntok * Dk);
  f2bf_kernel<<<dim3(2048), eb, 0, stream>>>(x_resid, Xr, (size_t)Ntok * Dk);
  biasc_kernel<<<dim3((Ftot + 255) / 256), eb, 0, stream>>>(b_dec_up, W_enc_down, biasc);

  // --- WMMA GEMMs ---
  wmma_gemm_kernel<0><<<dim3(Ftot / 128, Ntok / 128), eb, 0, stream>>>(
      Xc, Wup_bf, up_pre, nullptr, b_enc_up, nullptr);
  wmma_gemm_kernel<1><<<dim3(Ftot / 128, Ntok / 128), eb, 0, stream>>>(
      Xr, Wdn_bf, initial, nullptr, nullptr, nullptr);
  wmma_gemm_kernel<2><<<dim3(Ftot / 128, Ftot / 128), eb, 0, stream>>>(
      WdupT, Wdn_bf, nullptr, maskedT, nullptr, conn_mask);

  // --- sparsify masked virtual weights (1% density -> ~30x traffic cut) ---
  compact_kernel<<<dim3((Ftot + 255) / 256), eb, 0, stream>>>(maskedT, ell, counts);

  // --- sparse token path ---
  topk_kernel<<<dim3(Ntok), eb, 0, stream>>>(up_pre, up_idx, up_val);
  down_fused_kernel<<<dim3(Ntok), eb, 0, stream>>>(
      initial, ell, counts, up_idx, up_val, biasc, b_enc_down, WddT,
      b_dec_down, (float*)d_out);
}
